// Attention_25323127177507
// MI455X (gfx1250) — compile-verified, matrix operand order checked by reading, not run on hardware
//
#include <hip/hip_runtime.h>
#include <math.h>
#include <stdint.h>

// ---------------------------------------------------------------------------
// Multi-head attention for MI455X (gfx1250), bf16 WMMA path with CDNA5
// async memory->LDS staging (GLOBAL_LOAD_ASYNC_TO_LDS_B128 / s_wait_asynccnt).
//   B=4, S=2048, D_MODEL=1024, H=16, d_k=64
// Pipeline:
//   (1) fp32 -> bf16 conversion of X and the four weight matrices
//   (2) WMMA GEMM  Q/K/V = X @ W^T + b   (Q pre-scaled by 1/sqrt(dk)),
//       double-buffered LDS, async tile loads
//   (3) flash attention per (b,h) with online softmax, all matmuls on
//       v_wmma_f32_16x16x32_bf16; Q/K tiles staged via async loads
//   (4) WMMA GEMM  out = ctx @ Wout^T + bout  (fp32 output)
// ---------------------------------------------------------------------------

typedef __attribute__((ext_vector_type(16))) __bf16 v16bf;
typedef __attribute__((ext_vector_type(8)))  float  v8f;

typedef unsigned short u16;
typedef unsigned int   u32;

#define D_MODEL 1024
#define NHEADS  16
#define DK      64
#define BATCH   4
#define SEQ     2048
#define MTOT    (BATCH * SEQ)   // 8192

// POD 16-byte chunk (HIP's uint4 has non-trivial ctors -> unusable in unions)
struct __attribute__((aligned(16))) U128 { u32 x, y, z, w; };

union FragBF {
  U128  q[2];   // 32 bytes
  v16bf v;      // 16 x bf16
};

__device__ __forceinline__ u16 f2bf(float f) {
  // round-to-nearest-even fp32 -> bf16
  u32 x = __float_as_uint(f);
  u32 r = (x + 0x7FFFu + ((x >> 16) & 1u)) >> 16;
  return (u16)r;
}

__device__ __forceinline__ v8f zero8() {
  v8f z;
#pragma unroll
  for (int i = 0; i < 8; ++i) z[i] = 0.0f;
  return z;
}

// Async copy of 16 bytes global -> LDS (ASYNCcnt-tracked, no VGPR transit).
// lds_off: LDS byte address (low 32 bits of the generic pointer = DS offset,
// since the LDS aperture lives in addr[63:32] per CDNA5 aperture rules).
__device__ __forceinline__ void async_cp16(const u16* lds_dst, const u16* gsrc) {
  u32 lds_off = (u32)(uintptr_t)lds_dst;
  unsigned long long ga = (unsigned long long)(uintptr_t)gsrc;
  asm volatile("global_load_async_to_lds_b128 %0, %1, off"
               :: "v"(lds_off), "v"(ga)
               : "memory");
}

__device__ __forceinline__ void wait_async0() {
  asm volatile("s_wait_asynccnt 0" ::: "memory");
}

// Load one 16x32 bf16 fragment from a row-major LDS tile (A and B share the
// lane pattern when the B source is row-major over its N index), per CDNA5
// ISA 7.12.2:
//   lane l (0..15):  row = l,    K = 0..7  (V0-3) and K = 16..23 (V4-7)
//   lane l (16..31): row = l-16, K = 8..15 (V0-3) and K = 24..31 (V4-7)
__device__ __forceinline__ void ld_frag(const u16* base, int stride, int lane,
                                        FragBF& f) {
  int r  = lane & 15;
  int hk = (lane >> 4) << 3;           // 0 or 8
  const u16* p = base + r * stride + hk;
  f.q[0] = *(const U128*)(p);
  f.q[1] = *(const U128*)(p + 16);
}

__device__ __forceinline__ v8f wmma_bf16(const FragBF& a, const FragBF& b, v8f c) {
  return __builtin_amdgcn_wmma_f32_16x16x32_bf16(
      /*neg_a=*/false, a.v, /*neg_b=*/false, b.v,
      /*c_mod=*/(short)0, c, /*reuse_a=*/false, /*reuse_b=*/false);
}

// ---------------------------------------------------------------------------
// (1) fp32 -> bf16 conversion
// ---------------------------------------------------------------------------
__global__ void cvt_bf16_kernel(const float* __restrict__ src,
                                u16* __restrict__ dst, int n) {
  int i    = blockIdx.x * blockDim.x + threadIdx.x;
  int strd = gridDim.x * blockDim.x;
  for (; i < n; i += strd) dst[i] = f2bf(src[i]);
}

// ---------------------------------------------------------------------------
// (2)+(4) GEMM: Y[m][n] = sum_k A[m][k] * W[n][k] + bias[n]
//   A: [8192][1024] bf16 row-major, W: [1024][1024] bf16 row-major.
//   mode 0: write bf16 head-major [b][h][s][d], value scaled by 'scale'
//   mode 1: write fp32 row-major  [m][n]
// Block = 128x128 output, 8 wave32s in 4(M) x 2(N); each wave owns 2x4 WMMA
// tiles. K-steps of 32 through double-buffered LDS filled by async loads.
// ---------------------------------------------------------------------------
__device__ __forceinline__ void gemm_stage_async(
    u16* lA, u16* lB, const u16* __restrict__ A, const u16* __restrict__ W,
    int m0, int n0, int k0, int tid) {
#pragma unroll
  for (int i = 0; i < 2; ++i) {
    int c   = tid + i * 256;           // 0..511
    int row = c >> 2;
    int col = (c & 3) << 3;
    async_cp16(&lA[row * 32 + col],
               &A[(size_t)(m0 + row) * D_MODEL + k0 + col]);
    async_cp16(&lB[row * 32 + col],
               &W[(size_t)(n0 + row) * D_MODEL + k0 + col]);
  }
}

__global__ __launch_bounds__(256)
void gemm_bf16_kernel(const u16* __restrict__ A, const u16* __restrict__ W,
                      const float* __restrict__ bias,
                      u16* __restrict__ outH, float* __restrict__ outF,
                      int mode, float scale) {
  __shared__ __attribute__((aligned(16))) u16 lA[2][128 * 32];
  __shared__ __attribute__((aligned(16))) u16 lB[2][128 * 32];

  const int tid  = threadIdx.x;
  const int lane = tid & 31;
  const int wave = tid >> 5;
  const int wm   = wave & 3;   // 0..3 along M
  const int wn   = wave >> 2;  // 0..1 along N
  const int m0   = blockIdx.x * 128;
  const int n0   = blockIdx.y * 128;

  v8f acc[2][4];
#pragma unroll
  for (int mi = 0; mi < 2; ++mi)
#pragma unroll
    for (int ni = 0; ni < 4; ++ni) acc[mi][ni] = zero8();

  // prologue: async-fill buffer 0
  gemm_stage_async(lA[0], lB[0], A, W, m0, n0, 0, tid);

  for (int k0 = 0; k0 < D_MODEL; k0 += 32) {
    const int cur = (k0 >> 5) & 1;
    wait_async0();          // my async fills (incl. next-buffer from prev iter)
    __syncthreads();        // everyone's fills visible; prior reads retired
    if (k0 + 32 < D_MODEL)
      gemm_stage_async(lA[cur ^ 1], lB[cur ^ 1], A, W, m0, n0, k0 + 32, tid);

    FragBF a[2], b[4];
#pragma unroll
    for (int mi = 0; mi < 2; ++mi)
      ld_frag(&lA[cur][(wm * 32 + mi * 16) * 32], 32, lane, a[mi]);
#pragma unroll
    for (int ni = 0; ni < 4; ++ni)
      ld_frag(&lB[cur][(wn * 64 + ni * 16) * 32], 32, lane, b[ni]);
#pragma unroll
    for (int mi = 0; mi < 2; ++mi)
#pragma unroll
      for (int ni = 0; ni < 4; ++ni)
        acc[mi][ni] = wmma_bf16(a[mi], b[ni], acc[mi][ni]);
  }

  // epilogue; C layout: lane l, VGPR r -> M = r + 8*(l>>4), N = (l&15)
  const int hi = lane >> 4;
  const int lc = lane & 15;
#pragma unroll
  for (int mi = 0; mi < 2; ++mi) {
#pragma unroll
    for (int ni = 0; ni < 4; ++ni) {
      int n  = n0 + wn * 64 + ni * 16 + lc;
      float bv = bias[n];
#pragma unroll
      for (int r = 0; r < 8; ++r) {
        int m = m0 + wm * 32 + mi * 16 + r + 8 * hi;
        float val = (acc[mi][ni][r] + bv) * scale;
        if (mode == 0) {
          int b = m >> 11;        // /2048
          int s = m & 2047;
          int h = n >> 6;         // /64
          int d = n & 63;
          outH[(((size_t)(b * NHEADS + h) * SEQ + s) * DK) + d] = f2bf(val);
        } else {
          outF[(size_t)m * D_MODEL + n] = val;
        }
      }
    }
  }
}

// ---------------------------------------------------------------------------
// (3) Flash attention. Q is pre-scaled by 1/sqrt(dk).
// Block: 256 threads (8 waves), one (b,h) head slice, 128 query rows.
// Q/K tiles staged by async loads; V staged transposed (register transit).
// Static LDS: 16+16+16+32 = 80 KB (< 320 KB/WGP on CDNA5).
// ---------------------------------------------------------------------------
__global__ __launch_bounds__(256)
void flash_attn_kernel(const u16* __restrict__ Qb, const u16* __restrict__ Kb,
                       const u16* __restrict__ Vb, u16* __restrict__ ctxb) {
  __shared__ __attribute__((aligned(16))) u16 lQ[128 * 64];
  __shared__ __attribute__((aligned(16))) u16 lK[128 * 64];
  __shared__ __attribute__((aligned(16))) u16 lVt[64 * 128];
  __shared__ __attribute__((aligned(16))) u16 lP[8 * 16 * 128];

  const int tid  = threadIdx.x;
  const int lane = tid & 31;
  const int wave = tid >> 5;
  const int hi   = lane >> 4;
  const int lc   = lane & 15;
  const int bh   = blockIdx.y;          // 0..63  (= b*16 + h)
  const int q0   = blockIdx.x * 128;    // query tile base
  const size_t headBase = (size_t)bh * SEQ * DK;

  // stage the 128x64 Q tile once (async; completion covered by the first
  // wait_async0 + barrier inside the kt loop)
#pragma unroll
  for (int i = 0; i < 4; ++i) {
    int c   = tid + i * 256;           // 0..1023
    int row = c >> 3;
    int col = (c & 7) << 3;
    async_cp16(&lQ[row * 64 + col],
               &Qb[headBase + (size_t)(q0 + row) * DK + col]);
  }

  float m_run[8], l_run[8];
  v8f   o[4];
#pragma unroll
  for (int r = 0; r < 8; ++r) { m_run[r] = -1.0e30f; l_run[r] = 0.0f; }
#pragma unroll
  for (int nt = 0; nt < 4; ++nt) o[nt] = zero8();

  u16* myP = &lP[wave * 16 * 128];

  for (int kt = 0; kt < 16; ++kt) {
    __syncthreads();                   // prior iteration's lK/lVt reads done
    const int kb = kt * 128;
    // K tile: async memory->LDS. V tile: sync load + transpose scatter.
#pragma unroll
    for (int i = 0; i < 4; ++i) {
      int c   = tid + i * 256;         // 0..1023
      int row = c >> 3;                // key index within tile
      int col = (c & 7) << 3;          // dk base
      async_cp16(&lK[row * 64 + col],
                 &Kb[headBase + (size_t)(kb + row) * DK + col]);
      U128 vv = *(const U128*)&Vb[headBase + (size_t)(kb + row) * DK + col];
      const u16* pv = (const u16*)&vv;
#pragma unroll
      for (int j = 0; j < 8; ++j) lVt[(col + j) * 128 + row] = pv[j];
    }
    wait_async0();
    __syncthreads();

    // ---- scores: S = Q(16x64) @ K^T(64x128), per wave ----
    FragBF aQ[2];
    ld_frag(&lQ[(wave * 16) * 64 + 0], 64, lane, aQ[0]);
    ld_frag(&lQ[(wave * 16) * 64 + 32], 64, lane, aQ[1]);

    v8f sc[8];
#pragma unroll
    for (int ni = 0; ni < 8; ++ni) {
      v8f s = zero8();
      FragBF bK;
      ld_frag(&lK[(ni * 16) * 64 + 0], 64, lane, bK);
      s = wmma_bf16(aQ[0], bK, s);
      ld_frag(&lK[(ni * 16) * 64 + 32], 64, lane, bK);
      s = wmma_bf16(aQ[1], bK, s);
      sc[ni] = s;
    }

    // ---- online softmax (rows live across the 16 lanes of each half) ----
    float corr[8];
#pragma unroll
    for (int r = 0; r < 8; ++r) {
      float mx = sc[0][r];
#pragma unroll
      for (int ni = 1; ni < 8; ++ni) mx = fmaxf(mx, sc[ni][r]);
#pragma unroll
      for (int off = 1; off < 16; off <<= 1)
        mx = fmaxf(mx, __shfl_xor(mx, off, 16));
      float mnew = fmaxf(m_run[r], mx);
      corr[r]  = __expf(m_run[r] - mnew);
      m_run[r] = mnew;
      float rs = 0.0f;
#pragma unroll
      for (int ni = 0; ni < 8; ++ni) {
        float p = __expf(sc[ni][r] - mnew);
        sc[ni][r] = p;
        rs += p;
      }
#pragma unroll
      for (int off = 1; off < 16; off <<= 1) rs += __shfl_xor(rs, off, 16);
      l_run[r] = l_run[r] * corr[r] + rs;
    }
#pragma unroll
    for (int nt = 0; nt < 4; ++nt)
#pragma unroll
      for (int r = 0; r < 8; ++r) o[nt][r] *= corr[r];

    // ---- spill P as bf16 into the wave-private LDS strip (A relayout) ----
#pragma unroll
    for (int ni = 0; ni < 8; ++ni)
#pragma unroll
      for (int r = 0; r < 8; ++r)
        myP[(r + 8 * hi) * 128 + ni * 16 + lc] = f2bf(sc[ni][r]);

    // ---- O += P(16x128) @ V(128x64) ----
#pragma unroll
    for (int kk = 0; kk < 4; ++kk) {
      FragBF aP;
      ld_frag(&myP[kk * 32], 128, lane, aP);
#pragma unroll
      for (int nt = 0; nt < 4; ++nt) {
        FragBF bV;
        ld_frag(&lVt[(nt * 16) * 128 + kk * 32], 128, lane, bV);
        o[nt] = wmma_bf16(aP, bV, o[nt]);
      }
    }
  }

  // ---- normalize and write ctx in concat-head layout [b][s][h*64+d] ----
  const int b = bh >> 4;
  const int h = bh & 15;
#pragma unroll
  for (int r = 0; r < 8; ++r) {
    float inv = 1.0f / l_run[r];
    int   s   = q0 + wave * 16 + r + 8 * hi;
    size_t rowOff = ((size_t)b * SEQ + s) * D_MODEL + h * DK;
#pragma unroll
    for (int nt = 0; nt < 4; ++nt) {
      int d = nt * 16 + lc;
      ctxb[rowOff + d] = f2bf(o[nt][r] * inv);
    }
  }
}

// ---------------------------------------------------------------------------
// launch
// ---------------------------------------------------------------------------
extern "C" void kernel_launch(void* const* d_in, const int* in_sizes, int n_in,
                              void* d_out, int out_size, void* d_ws,
                              size_t ws_size, hipStream_t stream) {
  const float* X  = (const float*)d_in[0];
  const float* Wq = (const float*)d_in[1];
  const float* bq = (const float*)d_in[2];
  const float* Wk = (const float*)d_in[3];
  const float* bk = (const float*)d_in[4];
  const float* Wv = (const float*)d_in[5];
  const float* bv = (const float*)d_in[6];
  const float* Wo = (const float*)d_in[7];
  const float* bo = (const float*)d_in[8];
  float* out = (float*)d_out;

  u16*   ws  = (u16*)d_ws;
  size_t off = 0;
  u16* Xb  = ws + off; off += (size_t)MTOT * D_MODEL;      // 16 MB
  u16* Wqb = ws + off; off += (size_t)D_MODEL * D_MODEL;   //  2 MB
  u16* Wkb = ws + off; off += (size_t)D_MODEL * D_MODEL;
  u16* Wvb = ws + off; off += (size_t)D_MODEL * D_MODEL;
  u16* Wob = ws + off; off += (size_t)D_MODEL * D_MODEL;
  u16* Qb  = ws + off; off += (size_t)MTOT * D_MODEL;      // head-major
  u16* Kb  = ws + off; off += (size_t)MTOT * D_MODEL;
  u16* Vb  = ws + off; off += (size_t)MTOT * D_MODEL;
  u16* Cb  = ws + off; off += (size_t)MTOT * D_MODEL;      // ctx, concat heads
  (void)ws_size; (void)in_sizes; (void)n_in; (void)out_size;

  const float qscale = 0.125f;  // 1/sqrt(64)

  cvt_bf16_kernel<<<2048, 256, 0, stream>>>(X, Xb, MTOT * D_MODEL);
  cvt_bf16_kernel<<<512, 256, 0, stream>>>(Wq, Wqb, D_MODEL * D_MODEL);
  cvt_bf16_kernel<<<512, 256, 0, stream>>>(Wk, Wkb, D_MODEL * D_MODEL);
  cvt_bf16_kernel<<<512, 256, 0, stream>>>(Wv, Wvb, D_MODEL * D_MODEL);
  cvt_bf16_kernel<<<512, 256, 0, stream>>>(Wo, Wob, D_MODEL * D_MODEL);

  dim3 gg(MTOT / 128, D_MODEL / 128);  // 64 x 8
  gemm_bf16_kernel<<<gg, 256, 0, stream>>>(Xb, Wqb, bq, Qb, nullptr, 0, qscale);
  gemm_bf16_kernel<<<gg, 256, 0, stream>>>(Xb, Wkb, bk, Kb, nullptr, 0, 1.0f);
  gemm_bf16_kernel<<<gg, 256, 0, stream>>>(Xb, Wvb, bv, Vb, nullptr, 0, 1.0f);

  flash_attn_kernel<<<dim3(SEQ / 128, BATCH * NHEADS), 256, 0, stream>>>(
      Qb, Kb, Vb, Cb);

  gemm_bf16_kernel<<<gg, 256, 0, stream>>>(Cb, Wob, bo, nullptr, out, 1, 1.0f);
}
